// GraphNeuralNetwork_37907381354913
// MI455X (gfx1250) — matrix operand order repaired and checked by reading, not compiled, not run on hardware
//
#include <hip/hip_runtime.h>
#include <stdint.h>

// ---------------- problem constants (match reference) ----------------
#define BATCH   8
#define NNODES  2048
#define DIN     32
#define HDIM    64
#define DOUT    32
#define KNN     8
#define ALPHA   0.2f
#define LN_EPS  1e-5f
#define NW      (NNODES / 32)      // adjacency words per row = 64

typedef __attribute__((ext_vector_type(16))) _Float16 v16h;
typedef __attribute__((ext_vector_type(8)))  float    v8f;

// ---------------- WMMA fragment loaders (contiguous k order, same in A and B) ----------------
// A: 16x32 f16, row-major source A[(row0+m)*lda + k0 + k]
//   lane L: m = L%16, g = L/16; vgpr r holds K = 16g + 2r, +1
__device__ __forceinline__ v16h load_a_frag(const float* __restrict__ A,
                                            int row0, int k0, int lda) {
  const int lane = threadIdx.x & 31;
  const int m = lane & 15, g = lane >> 4;
  const float* p = A + (size_t)(row0 + m) * lda + k0;
  v16h a;
#pragma unroll
  for (int r = 0; r < 8; ++r) {
    const int kb = 16 * g + 2 * r;
    a[2 * r]     = (_Float16)p[kb];
    a[2 * r + 1] = (_Float16)p[kb + 1];
  }
  return a;
}

// B: 32x16 f16, source W[(k0+k)*ldb + n0 + n]
//   lane L: n = L%16, g = L/16; vgpr r holds K = 16g + 2r, +1
__device__ __forceinline__ v16h load_b_frag(const float* __restrict__ W,
                                            int k0, int n0, int ldb) {
  const int lane = threadIdx.x & 31;
  const int n = lane & 15, g = lane >> 4;
  v16h b;
#pragma unroll
  for (int r = 0; r < 8; ++r) {
    const int k = 16 * g + 2 * r;
    b[2 * r]     = (_Float16)W[(size_t)(k0 + k) * ldb + n0 + n];
    b[2 * r + 1] = (_Float16)W[(size_t)(k0 + k + 1) * ldb + n0 + n];
  }
  return b;
}

// B = Z^T (Gram): B[k][n] = Z[(j0+n)*ld + k]  -> just a row load of z_j
__device__ __forceinline__ v16h load_bT_frag(const float* __restrict__ Z,
                                             int j0, int ld) {
  const int lane = threadIdx.x & 31;
  const int n = lane & 15, g = lane >> 4;
  const float* p = Z + (size_t)(j0 + n) * ld;
  v16h b;
#pragma unroll
  for (int r = 0; r < 8; ++r) {
    const int k = 16 * g + 2 * r;
    b[2 * r]     = (_Float16)p[k];
    b[2 * r + 1] = (_Float16)p[k + 1];
  }
  return b;
}

__device__ __forceinline__ v8f wmma_f16(v16h a, v16h b, v8f c) {
  return __builtin_amdgcn_wmma_f32_16x16x32_f16(false, a, false, b,
                                                (short)0, c, false, false);
}

// ---------------- kernel 1: row sum-of-squares ----------------
__global__ void gnn_sq_kernel(const float* __restrict__ z, float* __restrict__ sq) {
  int r = blockIdx.x * blockDim.x + threadIdx.x;
  if (r >= BATCH * NNODES) return;
  const float* p = z + (size_t)r * DIN;
  float s = 0.f;
#pragma unroll
  for (int f = 0; f < DIN; ++f) s += p[f] * p[f];
  sq[r] = s;
}

// ---------------- kernel 2: fused WMMA Gram + top-8 kNN ----------------
// one wave per 16-row tile; never materializes the NxN distance matrix.
__global__ __launch_bounds__(32)
void gnn_knn_kernel(const float* __restrict__ z, const float* __restrict__ sq,
                    int* __restrict__ knn) {
  const int b  = blockIdx.x / (NNODES / 16);
  const int i0 = (blockIdx.x % (NNODES / 16)) * 16;
  const float* zb  = z  + (size_t)b * NNODES * DIN;
  const float* sqb = sq + (size_t)b * NNODES;

  const int lane = threadIdx.x;
  const int n = lane & 15, g = lane >> 4;

  const v16h afrag = load_a_frag(zb, i0, 0, DIN);   // 16x32 z rows, loaded once

  __shared__ float tile[16 * 17];                   // padded vs bank conflicts

  // per-row top-8 (lanes 0..15 own rows)
  float bd[KNN]; int bi[KNN];
#pragma unroll
  for (int q = 0; q < KNN; ++q) { bd[q] = 3.0e38f; bi[q] = 0; }
  const int myrow = i0 + lane;                      // valid for lanes 0..15

  for (int jt = 0; jt < NNODES / 16; ++jt) {
    const int j0 = jt * 16;
    v16h bfrag = load_bT_frag(zb, j0, DIN);
    v8f c = {};
    c = wmma_f16(afrag, bfrag, c);                  // c[r] = z_i(r+8g) . z_j(n)
#pragma unroll
    for (int r = 0; r < 8; ++r)
      tile[(r + 8 * g) * 17 + n] = c[r];
    __syncthreads();

    if (lane < 16) {
      const float* trow = &tile[lane * 17];
#pragma unroll
      for (int t = 0; t < 16; ++t) {
        const int j = j0 + t;
        // ranking: sq_i is row-constant -> dropped; self excluded (ref +1e6)
        const float dist = sqb[j] - 2.0f * trow[t];
        if (j != myrow && dist < bd[KNN - 1]) {
          bd[KNN - 1] = dist; bi[KNN - 1] = j;
#pragma unroll
          for (int q = KNN - 1; q > 0; --q)
            if (bd[q] < bd[q - 1]) {
              float td = bd[q]; bd[q] = bd[q - 1]; bd[q - 1] = td;
              int   tj = bi[q]; bi[q] = bi[q - 1]; bi[q - 1] = tj;
            }
        }
      }
    }
    __syncthreads();
  }

  if (lane < 16) {
    int* out = knn + ((size_t)b * NNODES + myrow) * KNN;
#pragma unroll
    for (int q = 0; q < KNN; ++q) out[q] = bi[q];
  }
}

// ---------------- kernel 3: scatter symmetric adjacency bitmask ----------------
__global__ void gnn_scatter_adj(const int* __restrict__ knn, uint32_t* __restrict__ adj) {
  int e = blockIdx.x * blockDim.x + threadIdx.x;
  if (e >= BATCH * NNODES * KNN) return;
  const int row = e / KNN;                 // b*N + i
  const int b = row / NNODES, i = row % NNODES;
  const int j = knn[e];
  atomicOr(&adj[(size_t)row * NW + (j >> 5)], 1u << (j & 31));
  atomicOr(&adj[((size_t)b * NNODES + j) * NW + (i >> 5)], 1u << (i & 31));
}

// ---------------- kernel 4: generic WMMA matmul C = A@W (+bias) ----------------
template <int KD, int NOUT, bool BIAS>
__global__ __launch_bounds__(32)
void gnn_mm_kernel(const float* __restrict__ A, const float* __restrict__ W,
                   const float* __restrict__ bias, float* __restrict__ C) {
  const int mt = blockIdx.x, nt = blockIdx.y;
  v8f c = {};
#pragma unroll
  for (int k0 = 0; k0 < KD; k0 += 32) {
    v16h a = load_a_frag(A, mt * 16, k0, KD);
    v16h b = load_b_frag(W, k0, nt * 16, NOUT);
    c = wmma_f16(a, b, c);
  }
  const int lane = threadIdx.x;
  const int n = lane & 15, g = lane >> 4;
  const float bb = BIAS ? bias[nt * 16 + n] : 0.f;
#pragma unroll
  for (int r = 0; r < 8; ++r)
    C[(size_t)(mt * 16 + r + 8 * g) * NOUT + nt * 16 + n] = c[r] + bb;
}

// ---------------- kernel 5: attention logit halves s = Wh.a_src, d = Wh.a_dst ---
template <int F>
__global__ void gnn_sd_kernel(const float* __restrict__ Wh,
                              const float* __restrict__ a_src,
                              const float* __restrict__ a_dst,
                              float* __restrict__ s, float* __restrict__ d) {
  int r = blockIdx.x * blockDim.x + threadIdx.x;
  if (r >= BATCH * NNODES) return;
  const float* p = Wh + (size_t)r * F;
  float ss = 0.f, dd = 0.f;
#pragma unroll
  for (int f = 0; f < F; ++f) { ss += p[f] * a_src[f]; dd += p[f] * a_dst[f]; }
  s[r] = ss; d[r] = dd;
}

__device__ __forceinline__ float gnn_wave_sum(float v) {
#pragma unroll
  for (int o = 16; o; o >>= 1) v += __shfl_xor(v, o, 32);
  return v;
}
__device__ __forceinline__ float gnn_wave_max(float v) {
#pragma unroll
  for (int o = 16; o; o >>= 1) v = fmaxf(v, __shfl_xor(v, o, 32));
  return v;
}

// ---------------- kernel 6: sparse softmax-aggregate + LayerNorm (+ReLU+res) ---
// one wave per node; neighbors from the 4MB L2-resident bitmask.
template <int F, bool RELU_RES>
__global__ __launch_bounds__(32)
void gnn_attn_kernel(const float* __restrict__ Wh, const float* __restrict__ s,
                     const float* __restrict__ dv, const uint32_t* __restrict__ adj,
                     const float* __restrict__ resid,
                     const float* __restrict__ gamma, const float* __restrict__ beta,
                     float* __restrict__ out) {
  const int row  = blockIdx.x;                 // b*N + i
  const int base = (row / NNODES) * NNODES;    // b*N
  const int lane = threadIdx.x;
  const uint32_t* arow = adj + (size_t)row * NW;
  const float si = s[row];

  // pass 1: max logit over neighbors (lanes split the bitmask words)
  float m = -3.0e38f;
  for (int w = lane; w < NW; w += 32) {
    uint32_t bits = arow[w];
    while (bits) {
      const int t = __builtin_ctz(bits); bits &= bits - 1;
      float e = si + dv[base + w * 32 + t];
      e = e >= 0.f ? e : ALPHA * e;
      m = fmaxf(m, e);
    }
  }
  m = gnn_wave_max(m);

  // pass 2: exp-weighted aggregation; lane owns channels f = lane (+32)
  constexpr int CPL = F / 32;
  float acc[CPL];
#pragma unroll
  for (int c = 0; c < CPL; ++c) acc[c] = 0.f;
  float lsum = 0.f;
  const float* Whb = Wh + (size_t)base * F;
  for (int w = 0; w < NW; ++w) {
    uint32_t bits = arow[w];
    while (bits) {
      const int t = __builtin_ctz(bits); bits &= bits - 1;
      const int j = w * 32 + t;
      float e = si + dv[base + j];
      e = e >= 0.f ? e : ALPHA * e;
      const float p = __expf(e - m);
      lsum += p;
#pragma unroll
      for (int c = 0; c < CPL; ++c)
        acc[c] += p * Whb[(size_t)j * F + lane + 32 * c];
    }
  }
  const float inv_l = 1.f / lsum;
#pragma unroll
  for (int c = 0; c < CPL; ++c) acc[c] *= inv_l;

  // fused LayerNorm across the F channels held by the wave
  float loc = 0.f;
#pragma unroll
  for (int c = 0; c < CPL; ++c) loc += acc[c];
  const float mean = gnn_wave_sum(loc) / F;
  float vloc = 0.f;
#pragma unroll
  for (int c = 0; c < CPL; ++c) vloc += (acc[c] - mean) * (acc[c] - mean);
  const float inv = rsqrtf(gnn_wave_sum(vloc) / F + LN_EPS);

#pragma unroll
  for (int c = 0; c < CPL; ++c) {
    const int f = lane + 32 * c;
    float y = (acc[c] - mean) * inv * gamma[f] + beta[f];
    if (RELU_RES) y = fmaxf(y, 0.f) + resid[(size_t)row * F + f];
    out[(size_t)row * F + f] = y;
  }
}

// ---------------- launch ----------------
extern "C" void kernel_launch(void* const* d_in, const int* in_sizes, int n_in,
                              void* d_out, int out_size, void* d_ws, size_t ws_size,
                              hipStream_t stream) {
  const float* z      = (const float*)d_in[0];
  const float* W_in   = (const float*)d_in[1];
  const float* b_in   = (const float*)d_in[2];
  const float* W1     = (const float*)d_in[3];
  const float* a1_src = (const float*)d_in[4];
  const float* a1_dst = (const float*)d_in[5];
  const float* g1     = (const float*)d_in[6];
  const float* be1    = (const float*)d_in[7];
  const float* W2     = (const float*)d_in[8];
  const float* a2_src = (const float*)d_in[9];
  const float* a2_dst = (const float*)d_in[10];
  const float* g2     = (const float*)d_in[11];
  const float* be2    = (const float*)d_in[12];
  float* out = (float*)d_out;

  const int BN = BATCH * NNODES;
  char* ws = (char*)d_ws;
  size_t off = 0;
  auto carve = [&](size_t bytes) { char* p = ws + off; off += (bytes + 255) & ~size_t(255); return p; };
  float*    sq   = (float*)   carve(sizeof(float) * BN);
  float*    sbuf = (float*)   carve(sizeof(float) * BN);
  float*    dbuf = (float*)   carve(sizeof(float) * BN);
  int*      knn  = (int*)     carve(sizeof(int) * BN * KNN);
  uint32_t* adj  = (uint32_t*)carve(sizeof(uint32_t) * (size_t)BN * NW);  // 4 MB
  float*    x0   = (float*)   carve(sizeof(float) * (size_t)BN * HDIM);
  float*    wh   = (float*)   carve(sizeof(float) * (size_t)BN * HDIM);
  float*    x1   = (float*)   carve(sizeof(float) * (size_t)BN * HDIM);

  // 1. graph construction (fused WMMA Gram + top-k; no NxN materialization)
  gnn_sq_kernel<<<(BN + 255) / 256, 256, 0, stream>>>(z, sq);
  gnn_knn_kernel<<<BATCH * (NNODES / 16), 32, 0, stream>>>(z, sq, knn);
  hipMemsetAsync(adj, 0, sizeof(uint32_t) * (size_t)BN * NW, stream);
  gnn_scatter_adj<<<(BN * KNN + 255) / 256, 256, 0, stream>>>(knn, adj);

  // 2. input projection: x0 = z @ W_in + b_in          [BN,32]@[32,64]
  gnn_mm_kernel<DIN, HDIM, true>
      <<<dim3(BN / 16, HDIM / 16), 32, 0, stream>>>(z, W_in, b_in, x0);

  // 3. layer 0: GAT -> LN -> ReLU -> +residual
  gnn_mm_kernel<HDIM, HDIM, false>
      <<<dim3(BN / 16, HDIM / 16), 32, 0, stream>>>(x0, W1, nullptr, wh);
  gnn_sd_kernel<HDIM><<<(BN + 255) / 256, 256, 0, stream>>>(wh, a1_src, a1_dst, sbuf, dbuf);
  gnn_attn_kernel<HDIM, true>
      <<<BN, 32, 0, stream>>>(wh, sbuf, dbuf, adj, x0, g1, be1, x1);

  // 4. layer 1: GAT -> LN (64 -> 32, no residual), write d_out
  gnn_mm_kernel<HDIM, DOUT, false>
      <<<dim3(BN / 16, DOUT / 16), 32, 0, stream>>>(x1, W2, nullptr, wh);
  gnn_sd_kernel<DOUT><<<(BN + 255) / 256, 256, 0, stream>>>(wh, a2_src, a2_dst, sbuf, dbuf);
  gnn_attn_kernel<DOUT, false>
      <<<BN, 32, 0, stream>>>(wh, sbuf, dbuf, adj, nullptr, g2, be2, out);
}
